// ScaledDotProductAttention_3118146256966
// MI455X (gfx1250) — compile-verified, matrix operand order checked by reading, not run on hardware
//
#include <hip/hip_runtime.h>

// B=4 H=16 S=2048 D=64; d_in = {key, query, value, mask}; d_out = output
// [B,H,S,D] f32 followed by attention_weight [B,H,S,S] f32.

typedef __attribute__((ext_vector_type(16))) _Float16     v16h;
typedef __attribute__((ext_vector_type(8)))  float        v8f;
typedef __attribute__((ext_vector_type(4)))  float        v4f;
typedef __attribute__((ext_vector_type(4)))  unsigned int v4u;
typedef __attribute__((ext_vector_type(8)))  int          v8i;
typedef __attribute__((ext_vector_type(4)))  int          v4i;

#define SEQ     2048
#define DH      64
#define QT      16          // query rows per workgroup
#define NWAVE   4           // 128 threads
#define LSTRIDE 2052        // 2048 + 4 pad (keeps 16B alignment, dodges banks)
#define VROWS   64          // V rows staged per TDM chunk (16 KB)

#if defined(__has_builtin)
#if __has_builtin(__builtin_amdgcn_tensor_load_to_lds) && \
    __has_builtin(__builtin_amdgcn_s_wait_tensorcnt)
#define USE_TDM 1
#endif
#endif
#ifndef USE_TDM
#define USE_TDM 0
#endif

#if USE_TDM
typedef __attribute__((address_space(3))) const void lds_cv;
static __device__ __forceinline__ unsigned lds_byte_offset(const void* p) {
    return (unsigned)(unsigned long long)(lds_cv*)p;   // generic -> LDS offset
}
#endif

// Fill a v16h in WMMA 16-bit A/B register layout from a 64-f32 row:
// lane-half `sel` owns K = off+sel*8..+7 and K = off+16+sel*8..+7.
#define FILL16(dst, src, off) do {                                          \
    v4f t0 = *(const v4f*)((src) + (off) + 0);                              \
    v4f t1 = *(const v4f*)((src) + (off) + 4);                              \
    v4f t2 = *(const v4f*)((src) + (off) + 16);                             \
    v4f t3 = *(const v4f*)((src) + (off) + 20);                             \
    (dst)[0]=(_Float16)t0.x;  (dst)[1]=(_Float16)t0.y;                      \
    (dst)[2]=(_Float16)t0.z;  (dst)[3]=(_Float16)t0.w;                      \
    (dst)[4]=(_Float16)t1.x;  (dst)[5]=(_Float16)t1.y;                      \
    (dst)[6]=(_Float16)t1.z;  (dst)[7]=(_Float16)t1.w;                      \
    (dst)[8]=(_Float16)t2.x;  (dst)[9]=(_Float16)t2.y;                      \
    (dst)[10]=(_Float16)t2.z; (dst)[11]=(_Float16)t2.w;                     \
    (dst)[12]=(_Float16)t3.x; (dst)[13]=(_Float16)t3.y;                     \
    (dst)[14]=(_Float16)t3.z; (dst)[15]=(_Float16)t3.w;                     \
} while (0)

__global__ __launch_bounds__(128, 1)
void attn_fused_kernel(const float* __restrict__ K,
                       const float* __restrict__ Q,
                       const float* __restrict__ V,
                       const unsigned char* __restrict__ mask,  // [S,S], 1=masked
                       float* __restrict__ out,                 // [BH,S,DH]
                       float* __restrict__ attnw)               // [BH,S,S]
{
    __shared__ float sc[QT * LSTRIDE];   // scores -> exp(scores - rowmax)
    __shared__ float rowinv[QT];         // 1 / rowsum
#if USE_TDM
    __shared__ float vstage[VROWS * DH]; // 16 KB V staging (TDM target)
#endif

    const int tid  = threadIdx.x;
    const int lane = tid & 31;
    const int wv   = tid >> 5;
    const int col  = lane & 15;
    const int sel  = lane >> 4;
    const int sel8 = sel * 8;

    const int bh    = blockIdx.x >> 7;   // 128 q-tiles per (b,h)
    const int qt    = blockIdx.x & 127;
    const int qbase = qt * QT;

    const float* Qb = Q + (size_t)bh * SEQ * DH;
    const float* Kb = K + (size_t)bh * SEQ * DH;
    const float* Vb = V + (size_t)bh * SEQ * DH;

    // ---- Q tile (16x64) into A-layout f16 ----
    v16h a_lo, a_hi;
    {
        const float* qr = Qb + (size_t)(qbase + col) * DH;
        FILL16(a_lo, qr, sel8);
        FILL16(a_hi, qr, 32 + sel8);
    }

    // ---- Phase 1: E = Q K^T, mask, scale, stage into LDS ----
    for (int t = wv; t < SEQ / 16; t += NWAVE) {
        const int kb = t * 16;
        const float* kr = Kb + (size_t)(kb + col) * DH;
        v16h b_lo, b_hi;
        FILL16(b_lo, kr, sel8);
        FILL16(b_hi, kr, 32 + sel8);

        v8f c = {};
        c = __builtin_amdgcn_wmma_f32_16x16x32_f16(false, a_lo, false, b_lo,
                                                   (short)0, c, false, false);
        c = __builtin_amdgcn_wmma_f32_16x16x32_f16(false, a_hi, false, b_hi,
                                                   (short)0, c, false, false);
        const int key = kb + col;
        const unsigned char* mrow = mask + (size_t)(qbase + sel8) * SEQ + key;
#pragma unroll
        for (int i = 0; i < 8; ++i) {
            const int r = i + sel8;                       // C row = i + 8*sel
            float x = mrow[(size_t)i * SEQ] ? -1e20f : c[i];
            sc[r * LSTRIDE + key] = x * 0.125f;           // 1/sqrt(64) after mask
        }
    }
    __syncthreads();

    // ---- Phase 2: row softmax (8 lanes/row, 16B LDS accesses) ----
    {
        const int r    = tid >> 3;
        const int sub4 = (tid & 7) * 4;
        float* rowp = &sc[r * LSTRIDE];

        v4f vmx = { -3.4e38f, -3.4e38f, -3.4e38f, -3.4e38f };
        for (int j = 0; j < SEQ / 32; ++j) {
            v4f x = *(const v4f*)(rowp + sub4 + 32 * j);
            vmx.x = fmaxf(vmx.x, x.x); vmx.y = fmaxf(vmx.y, x.y);
            vmx.z = fmaxf(vmx.z, x.z); vmx.w = fmaxf(vmx.w, x.w);
        }
        float mx = fmaxf(fmaxf(vmx.x, vmx.y), fmaxf(vmx.z, vmx.w));
        mx = fmaxf(mx, __shfl_xor(mx, 4, 8));
        mx = fmaxf(mx, __shfl_xor(mx, 2, 8));
        mx = fmaxf(mx, __shfl_xor(mx, 1, 8));

        float sum = 0.f;
        for (int j = 0; j < SEQ / 32; ++j) {
            v4f x = *(const v4f*)(rowp + sub4 + 32 * j);
            v4f e;
            e.x = __expf(x.x - mx); e.y = __expf(x.y - mx);
            e.z = __expf(x.z - mx); e.w = __expf(x.w - mx);
            *(v4f*)(rowp + sub4 + 32 * j) = e;            // unnormalized prob
            sum += e.x + e.y + e.z + e.w;
        }
        sum += __shfl_xor(sum, 4, 8);
        sum += __shfl_xor(sum, 2, 8);
        sum += __shfl_xor(sum, 1, 8);
        if ((tid & 7) == 0) rowinv[r] = 1.0f / sum;
    }
    __syncthreads();

    // ---- Phase 2b: stream normalized attention to HBM (NT b128 stores) ----
    {
        float* aw = attnw + (size_t)bh * SEQ * SEQ + (size_t)qbase * SEQ;
#pragma unroll 4
        for (int i = 0; i < (QT * SEQ) / (128 * 4); ++i) {
            const int idx4 = i * 128 + tid;
            const int r = idx4 >> 9;                      // (idx4*4) / 2048
            const int c = (idx4 * 4) & (SEQ - 1);
            v4f p = *(const v4f*)&sc[r * LSTRIDE + c];
            const float inv = rowinv[r];
            v4f o4 = { p.x * inv, p.y * inv, p.z * inv, p.w * inv };
            __builtin_nontemporal_store(o4, (v4f*)(aw + (size_t)r * SEQ + c));
        }
    }

    // ---- Phase 3: O = P V  (wave wv owns d-slice wv*16..+15) ----
    v8f o = {};
    const int dn = wv * 16 + col;

#if USE_TDM
    for (int ch = 0; ch < SEQ / VROWS; ++ch) {
        __syncthreads();                       // vstage free for overwrite
        if (wv == 0) {
            const unsigned long long ga =
                (unsigned long long)(const void*)(Vb + (size_t)ch * VROWS * DH);
            v4u g0;
            g0[0] = 1u;                                        // count=1
            g0[1] = lds_byte_offset(&vstage[0]);               // lds_addr
            g0[2] = (unsigned)(ga & 0xFFFFFFFFu);              // global_addr lo
            g0[3] = (unsigned)((ga >> 32) & 0x1FFFFFFu) | (2u << 30); // hi|type=2
            v8i g1;
            g1[0] = (int)(2u << 16);          // data_size = 4 bytes
            g1[1] = (int)((unsigned)DH << 16);// tensor_dim0 = 64
            g1[2] = (int)((unsigned)SEQ << 16);// dim0 hi=0 | tensor_dim1 lo = 2048
            g1[3] = (int)((unsigned)DH << 16);// dim1 hi=0 | tile_dim0 = 64
            g1[4] = (int)VROWS;               // tile_dim1 = 64, tile_dim2 = 0
            g1[5] = DH;                       // tensor_dim0_stride = 64
            g1[6] = 0;
            g1[7] = 0;
            v4i gz = { 0, 0, 0, 0 };
#if __clang_major__ >= 23
            v8i g5 = { 0, 0, 0, 0, 0, 0, 0, 0 };
            __builtin_amdgcn_tensor_load_to_lds(g0, g1, gz, gz, g5, 0);
#else
            __builtin_amdgcn_tensor_load_to_lds(g0, g1, gz, gz, 0);
#endif
            __builtin_amdgcn_s_wait_tensorcnt(0);
        }
        __syncthreads();                       // vstage ready for all waves
#pragma unroll
        for (int kk2 = 0; kk2 < VROWS / 32; ++kk2) {
            const int kbb  = ch * VROWS + kk2 * 32;
            const int lrow = kk2 * 32 + sel8;
            v16h pa, vb;
            const float* prow = &sc[col * LSTRIDE + kbb + sel8];
            v4f p0 = *(const v4f*)(prow);
            v4f p1 = *(const v4f*)(prow + 4);
            v4f p2 = *(const v4f*)(prow + 16);
            v4f p3 = *(const v4f*)(prow + 20);
            pa[0]=(_Float16)p0.x;  pa[1]=(_Float16)p0.y;
            pa[2]=(_Float16)p0.z;  pa[3]=(_Float16)p0.w;
            pa[4]=(_Float16)p1.x;  pa[5]=(_Float16)p1.y;
            pa[6]=(_Float16)p1.z;  pa[7]=(_Float16)p1.w;
            pa[8]=(_Float16)p2.x;  pa[9]=(_Float16)p2.y;
            pa[10]=(_Float16)p2.z; pa[11]=(_Float16)p2.w;
            pa[12]=(_Float16)p3.x; pa[13]=(_Float16)p3.y;
            pa[14]=(_Float16)p3.z; pa[15]=(_Float16)p3.w;
#pragma unroll
            for (int e = 0; e < 8; ++e) {
                vb[e]     = (_Float16)vstage[(lrow + e) * DH + dn];
                vb[e + 8] = (_Float16)vstage[(lrow + 16 + e) * DH + dn];
            }
            o = __builtin_amdgcn_wmma_f32_16x16x32_f16(false, pa, false, vb,
                                                       (short)0, o, false, false);
        }
    }
#else
    for (int kk = 0; kk < SEQ / 32; ++kk) {
        const int kbb = kk * 32;
        v16h pa, vb;
        const float* prow = &sc[col * LSTRIDE + kbb + sel8];
        v4f p0 = *(const v4f*)(prow);
        v4f p1 = *(const v4f*)(prow + 4);
        v4f p2 = *(const v4f*)(prow + 16);
        v4f p3 = *(const v4f*)(prow + 20);
        pa[0]=(_Float16)p0.x;  pa[1]=(_Float16)p0.y;
        pa[2]=(_Float16)p0.z;  pa[3]=(_Float16)p0.w;
        pa[4]=(_Float16)p1.x;  pa[5]=(_Float16)p1.y;
        pa[6]=(_Float16)p1.z;  pa[7]=(_Float16)p1.w;
        pa[8]=(_Float16)p2.x;  pa[9]=(_Float16)p2.y;
        pa[10]=(_Float16)p2.z; pa[11]=(_Float16)p2.w;
        pa[12]=(_Float16)p3.x; pa[13]=(_Float16)p3.y;
        pa[14]=(_Float16)p3.z; pa[15]=(_Float16)p3.w;
        const float* vcol = Vb + (size_t)(kbb + sel8) * DH + dn;
#pragma unroll
        for (int e = 0; e < 8; ++e) {
            vb[e]     = (_Float16)vcol[(size_t)e * DH];
            vb[e + 8] = (_Float16)vcol[(size_t)(16 + e) * DH];
        }
        o = __builtin_amdgcn_wmma_f32_16x16x32_f16(false, pa, false, vb,
                                                   (short)0, o, false, false);
    }
#endif

    {
        float* ob = out + (size_t)bh * SEQ * DH;
#pragma unroll
        for (int i = 0; i < 8; ++i) {
            const int r = i + sel8;
            ob[(size_t)(qbase + r) * DH + dn] = o[i] * rowinv[r];
        }
    }
}

extern "C" void kernel_launch(void* const* d_in, const int* in_sizes, int n_in,
                              void* d_out, int out_size, void* d_ws, size_t ws_size,
                              hipStream_t stream) {
    (void)in_sizes; (void)n_in; (void)d_ws; (void)ws_size; (void)out_size;
    const float* key   = (const float*)d_in[0];
    const float* query = (const float*)d_in[1];
    const float* value = (const float*)d_in[2];
    const unsigned char* mask = (const unsigned char*)d_in[3];  // jax bool = 1B

    const int BH = 4 * 16;
    float* out   = (float*)d_out;                               // [B,H,S,D]
    float* attnw = out + (size_t)BH * SEQ * DH;                 // [B,H,S,S]

    dim3 grid(BH * (SEQ / QT));   // 8192 workgroups
    dim3 block(128);              // 4 waves (wave32)
    attn_fused_kernel<<<grid, block, 0, stream>>>(key, query, value, mask,
                                                  out, attnw);
}
